// Inst3D_41386304864311
// MI455X (gfx1250) — matrix operand order repaired and checked by reading, not compiled, not run on hardware
//
#include <hip/hip_runtime.h>
#include <math.h>

// ---------------- problem constants ----------------
static constexpr int BB  = 8;
static constexpr int LL  = 1024;
static constexpr int DD  = 768;
static constexpr int HH  = 12;
static constexpr int DK  = 64;
static constexpr int SD  = 5;
static constexpr int WSZ = DD * DD;          // 589824 weights per matrix
static constexpr int NE  = BB * LL * DD;     // 6291456

typedef __attribute__((ext_vector_type(16))) _Float16 v16h;
typedef __attribute__((ext_vector_type(8)))  _Float16 v8h;
typedef __attribute__((ext_vector_type(8)))  float    v8f;

__device__ __forceinline__ v8f wmma_f16(v16h a, v16h b, v8f c) {
    return __builtin_amdgcn_wmma_f32_16x16x32_f16(
        /*neg_a=*/false, a, /*neg_b=*/false, b,
        /*c_mod=*/(short)0, c, /*reuse_a=*/false, /*reuse_b=*/false);
}

__device__ __forceinline__ void lds_wave_fence() {
    // cross-lane LDS RAW inside one wave: wait all DS ops, stop reordering
    asm volatile("s_wait_dscnt 0x0" ::: "memory");
}

// ---------------- kernel 1: transpose + convert weights to f16 ----------------
// wT[m][o*768 + i] = (f16) w_m[i*768 + o],  m in {q,k,v,fc}
__global__ void k_cvt_w(const float* __restrict__ wq, const float* __restrict__ wk,
                        const float* __restrict__ wv, const float* __restrict__ wfc,
                        _Float16* __restrict__ wT) {
    int idx = blockIdx.x * blockDim.x + threadIdx.x;
    if (idx >= 4 * WSZ) return;
    int m = idx / WSZ, r = idx % WSZ;
    int o = r / DD, i = r % DD;
    const float* src = (m == 0) ? wq : (m == 1) ? wk : (m == 2) ? wv : wfc;
    wT[(size_t)m * WSZ + (size_t)o * DD + i] = (_Float16)src[(size_t)i * DD + o];
}

// ---------------- kernel 2: QKV projection (WMMA) ----------------
// one wave per 16x16 output tile; K-loop 768/32 = 24 WMMAs.
// Q,K stored [h][b][l][dk] (dk contiguous); V stored transposed [h][b][dk][t].
__global__ __launch_bounds__(32)
void k_proj(const float* __restrict__ xq, const float* __restrict__ xk,
            const float* __restrict__ xv, const _Float16* __restrict__ wT,
            const float* __restrict__ bq, const float* __restrict__ bk,
            const float* __restrict__ bv,
            _Float16* __restrict__ Qh, _Float16* __restrict__ Kh,
            _Float16* __restrict__ VhT) {
    int tile  = blockIdx.x;                 // 3 * 512 * 48 tiles
    int which = tile / (512 * 48);
    int rt    = (tile / 48) % 512;
    int jt    = tile % 48;
    int lane  = threadIdx.x;
    int n  = lane & 15;
    int hi = lane >> 4;
    int klo = hi ? 8 : 0;

    const float*    X    = (which == 0) ? xq : (which == 1) ? xk : xv;
    const _Float16* W    = wT + (size_t)which * WSZ;
    const float*    bias = (which == 0) ? bq : (which == 1) ? bk : bv;

    int row  = rt * 16 + n;                 // A-matrix row for this lane
    int coln = jt * 16 + n;                 // B-matrix column for this lane
    const float*    xr = X + (size_t)row * DD;
    const _Float16* wr = W + (size_t)coln * DD;

    v8f c = {};
    for (int k0 = 0; k0 < DD; k0 += 32) {
        v16h a;
        #pragma unroll
        for (int i = 0; i < 8; ++i) a[i]     = (_Float16)xr[k0 + 2 * klo + i];
        #pragma unroll
        for (int i = 0; i < 8; ++i) a[8 + i] = (_Float16)xr[k0 + 2 * klo + 8 + i];
        v16h b = *(const v16h*)(wr + k0 + (hi ? 16 : 0));
        c = wmma_f16(a, b, c);
    }

    #pragma unroll
    for (int r = 0; r < 8; ++r) {
        int rr   = r + 8 * hi;
        int grow = rt * 16 + rr;            // global row = b*L + l
        int col  = jt * 16 + n;             // 0..767
        float v  = c[r] + bias[col];
        int bb = grow >> 10, l = grow & 1023;
        int h  = col >> 6,  dk = col & 63;
        size_t hb = (size_t)(h * BB + bb);
        if (which == 2) {
            VhT[(hb * DK + dk) * LL + l] = (_Float16)v;        // transposed V
        } else {
            _Float16* dst = (which == 0) ? Qh : Kh;
            dst[(hb * LL + l) * DK + dk] = (_Float16)v;
        }
    }
}

// ---------------- kernel 3: fused attention ----------------
// block = (batch b, 16-row tile l0); 12 waves = 12 heads.
// 32 keys per iteration: two QK^T c-tiles, one joint flash update, full-K PV.
// pairwise tile staged in LDS once and shared by all heads; logits spilled to
// the fused_attn output region and normalized in pass 2.
__global__ __launch_bounds__(384)
void k_attn(const _Float16* __restrict__ Qh, const _Float16* __restrict__ Kh,
            const _Float16* __restrict__ VhT, const float* __restrict__ ploc,
            const unsigned char* __restrict__ kpm, const float* __restrict__ wloc,
            const float* __restrict__ bloc, float* __restrict__ fused,
            _Float16* __restrict__ attnout) {
    __shared__ float ptile[16 * 32 * SD];        // 10240 B pairwise tile (32 keys)
    __shared__ float pscr[HH][16 * 33];          // per-wave P transpose scratch

    int h    = threadIdx.x >> 5;                 // head = wave id
    int lane = threadIdx.x & 31;
    int bb   = blockIdx.x >> 6;
    int l0   = (blockIdx.x & 63) << 4;
    int n    = lane & 15;
    int hi   = lane >> 4;
    int klo  = hi ? 8 : 0;

    size_t hb = (size_t)(h * BB + bb);
    const _Float16* Qp = Qh  + (hb * LL + l0) * DK;
    const _Float16* Kp = Kh  + hb * LL * DK;
    const _Float16* Vp = VhT + hb * DK * LL;

    float wl[SD];
    #pragma unroll
    for (int d = 0; d < SD; ++d) wl[d] = wloc[d * HH + h];
    float bl = bloc[h];

    // Q A-tiles for both K-steps (dk 0..31, 32..63)
    const _Float16* qr = Qp + (size_t)n * DK;
    union { v16h v; v8h p[2]; } a0, a1;
    a0.p[0] = *(const v8h*)(qr + 2 * klo);
    a0.p[1] = *(const v8h*)(qr + 2 * klo + 8);
    a1.p[0] = *(const v8h*)(qr + 2 * klo + 32);
    a1.p[1] = *(const v8h*)(qr + 2 * klo + 40);

    float mrow[8], srow[8];
    v8f O[4];
    #pragma unroll
    for (int r = 0; r < 8; ++r) { mrow[r] = -1e30f; srow[r] = 0.0f; }
    #pragma unroll
    for (int j = 0; j < 4; ++j) O[j] = v8f{};

    size_t fb0 = hb * (size_t)LL * LL;           // fused[h][b] base

    for (int tt2 = 0; tt2 < 32; ++tt2) {
        int t0 = tt2 * 32;

        __syncthreads();                          // previous tile fully consumed
        for (int i = threadIdx.x; i < 16 * 32 * SD; i += 384) {
            int pos = i / SD, d = i % SD;
            int prow = pos >> 5, pcol = pos & 31;
            ptile[i] = ploc[((size_t)(bb * LL + l0 + prow) * LL + (t0 + pcol)) * SD + d];
        }
        // prefetch next 32-key pairwise tile (16 rows x 640B) toward the WGP
        if (tt2 + 1 < 32 && threadIdx.x < 160) {
            int prow = threadIdx.x / 10, poff = (threadIdx.x % 10) * 16; // floats
            __builtin_prefetch(
                &ploc[((size_t)(bb * LL + l0 + prow) * LL + (t0 + 32)) * SD + poff],
                0, 3);
        }
        __syncthreads();

        // ---- Q . K^T: two 16-col tiles (4 WMMAs) ----
        const _Float16* krA = Kp + (size_t)(t0 + n) * DK + (hi ? 16 : 0);
        const _Float16* krB = krA + 16 * DK;
        v8f cA = {}, cB = {};
        cA = wmma_f16(a0.v, *(const v16h*)(krA),      cA);
        cA = wmma_f16(a1.v, *(const v16h*)(krA + 32), cA);
        cB = wmma_f16(a0.v, *(const v16h*)(krB),      cB);
        cB = wmma_f16(a1.v, *(const v16h*)(krB + 32), cB);

        bool maskedA = kpm[bb * LL + t0 + n] != 0;       // mask is column-only
        bool maskedB = kpm[bb * LL + t0 + 16 + n] != 0;

        // ---- logits = log(clip(relu(loc),1e-6)) + scale*attn ; spill ----
        float lgA[8], lgB[8];
        #pragma unroll
        for (int r = 0; r < 8; ++r) {
            int row = r + 8 * hi;
            const float* pvA = &ptile[(row * 32 + n) * SD];
            const float* pvB = pvA + 16 * SD;
            float dA = bl, dB = bl;
            #pragma unroll
            for (int d = 0; d < SD; ++d) { dA += pvA[d] * wl[d]; dB += pvB[d] * wl[d]; }
            float la = maskedA ? -INFINITY
                               : (__logf(fmaxf(fmaxf(dA, 0.0f), 1e-6f)) + 0.125f * cA[r]);
            float lb = maskedB ? -INFINITY
                               : (__logf(fmaxf(fmaxf(dB, 0.0f), 1e-6f)) + 0.125f * cB[r]);
            lgA[r] = la; lgB[r] = lb;
            size_t fr_ = fb0 + (size_t)(l0 + row) * LL + t0 + n;
            fused[fr_]      = la;                        // raw logits (pass 2 fixes)
            fused[fr_ + 16] = lb;
        }

        // ---- flash update over 32 cols: row max / sum (16-lane butterflies) ----
        float frv[8];
        #pragma unroll
        for (int r = 0; r < 8; ++r) {
            int row = r + 8 * hi;
            float mx = fmaxf(lgA[r], lgB[r]);
            #pragma unroll
            for (int o = 1; o < 16; o <<= 1) mx = fmaxf(mx, __shfl_xor(mx, o, 32));
            float mnew = fmaxf(mrow[r], mx);
            float fr   = __expf(mrow[r] - mnew);
            float pA   = __expf(lgA[r] - mnew);
            float pB   = __expf(lgB[r] - mnew);
            float ps   = pA + pB;
            #pragma unroll
            for (int o = 1; o < 16; o <<= 1) ps += __shfl_xor(ps, o, 32);
            srow[r] = srow[r] * fr + ps;
            mrow[r] = mnew;
            frv[r]  = fr;
            pscr[h][row * 33 + n]      = pA;             // c-layout -> LDS
            pscr[h][row * 33 + 16 + n] = pB;
        }
        lds_wave_fence();

        // ---- P tile (16x32) in A layout, full K ----
        union { v16h v; _Float16 e[16]; } pa;
        const float* sr = &pscr[h][(lane & 15) * 33];
        #pragma unroll
        for (int i = 0; i < 8; ++i) pa.e[i]     = (_Float16)sr[2 * klo + i];
        #pragma unroll
        for (int i = 0; i < 8; ++i) pa.e[8 + i] = (_Float16)sr[2 * klo + 8 + i];

        // ---- rescale O, accumulate P . V over 4 dk-tiles (full k=32) ----
        #pragma unroll
        for (int j = 0; j < 4; ++j) {
            #pragma unroll
            for (int r = 0; r < 8; ++r) O[j][r] *= frv[r];
            v16h vb = *(const v16h*)(Vp + (size_t)(j * 16 + n) * LL + t0 + (hi ? 16 : 0));
            O[j] = wmma_f16(pa.v, vb, O[j]);
        }
    }

    // ---- write attention output (B,L, h*64+dk) as f16 for FC ----
    #pragma unroll
    for (int j = 0; j < 4; ++j) {
        #pragma unroll
        for (int r = 0; r < 8; ++r) {
            int row = r + 8 * hi;
            float v = O[j][r] / srow[r];
            attnout[(size_t)(bb * LL + l0 + row) * DD + h * DK + j * 16 + n] =
                (_Float16)v;
        }
    }

    // ---- pass 2: logits -> probabilities in place ----
    for (int tt = 0; tt < 64; ++tt) {
        int t0 = tt * 16;
        #pragma unroll
        for (int r = 0; r < 8; ++r) {
            int row = r + 8 * hi;
            size_t idx = fb0 + (size_t)(l0 + row) * LL + t0 + n;
            float lgv = fused[idx];
            fused[idx] = __expf(lgv - mrow[r]) / srow[r];
        }
    }
}

// ---------------- kernel 4: FC (WMMA) + residual + layernorm ----------------
// block = 16 rows x full 768 cols; 8 waves, 6 col-tiles each; row stats in LDS.
__global__ __launch_bounds__(256)
void k_fc_ln(const _Float16* __restrict__ attnout, const _Float16* __restrict__ wfcT,
             const float* __restrict__ bfc, const float* __restrict__ resid,
             const float* __restrict__ lng, const float* __restrict__ lnb,
             float* __restrict__ out) {
    __shared__ float xbuf[16 * DD];              // 48 KB
    int wave = threadIdx.x >> 5, lane = threadIdx.x & 31;
    int n = lane & 15, hi = lane >> 4, klo = hi ? 8 : 0;
    int grow0 = blockIdx.x * 16;
    const _Float16* ar = attnout + (size_t)(grow0 + n) * DD;

    for (int jt = wave * 6; jt < wave * 6 + 6; ++jt) {
        int coln = jt * 16 + n;
        const _Float16* wr = wfcT + (size_t)coln * DD;
        v8f c = {};
        for (int k0 = 0; k0 < DD; k0 += 32) {
            union { v16h v; v8h p[2]; } a;
            a.p[0] = *(const v8h*)(ar + k0 + 2 * klo);
            a.p[1] = *(const v8h*)(ar + k0 + 2 * klo + 8);
            v16h b = *(const v16h*)(wr + k0 + (hi ? 16 : 0));
            c = wmma_f16(a.v, b, c);
        }
        #pragma unroll
        for (int r = 0; r < 8; ++r) {
            int rr = r + 8 * hi;
            int col = jt * 16 + n;
            float v = c[r] + bfc[col] + resid[(size_t)(grow0 + rr) * DD + col];
            xbuf[rr * DD + col] = v;
        }
    }
    __syncthreads();

    for (int rr = wave * 2; rr < wave * 2 + 2; ++rr) {
        float s = 0.0f, s2 = 0.0f;
        for (int c0 = lane; c0 < DD; c0 += 32) {
            float x = xbuf[rr * DD + c0];
            s += x; s2 += x * x;
        }
        #pragma unroll
        for (int o = 1; o < 32; o <<= 1) {
            s  += __shfl_xor(s,  o, 32);
            s2 += __shfl_xor(s2, o, 32);
        }
        float mean = s * (1.0f / DD);
        float var  = s2 * (1.0f / DD) - mean * mean;
        float inv  = rsqrtf(var + 1e-5f);
        for (int c0 = lane; c0 < DD; c0 += 32) {
            float x = xbuf[rr * DD + c0];
            out[(size_t)(grow0 + rr) * DD + c0] = (x - mean) * inv * lng[c0] + lnb[c0];
        }
    }
}

// ---------------- host launcher ----------------
extern "C" void kernel_launch(void* const* d_in, const int* in_sizes, int n_in,
                              void* d_out, int out_size, void* d_ws, size_t ws_size,
                              hipStream_t stream) {
    (void)in_sizes; (void)n_in; (void)out_size; (void)ws_size;
    const float* q    = (const float*)d_in[0];
    const float* k    = (const float*)d_in[1];
    const float* v    = (const float*)d_in[2];
    const float* ploc = (const float*)d_in[3];
    const unsigned char* kpm = (const unsigned char*)d_in[4];
    const float* wq = (const float*)d_in[5];
    const float* bq = (const float*)d_in[6];
    const float* wk = (const float*)d_in[7];
    const float* bk = (const float*)d_in[8];
    const float* wv = (const float*)d_in[9];
    const float* bv = (const float*)d_in[10];
    const float* wfc = (const float*)d_in[11];
    const float* bfc = (const float*)d_in[12];
    const float* wloc = (const float*)d_in[13];
    const float* bloc = (const float*)d_in[14];
    const float* lng = (const float*)d_in[15];
    const float* lnb = (const float*)d_in[16];

    float* out   = (float*)d_out;                 // (B,L,D) f32
    float* fused = out + (size_t)NE;              // (H,B,L,L) f32

    _Float16* wT      = (_Float16*)d_ws;          // 4*WSZ f16 transposed weights
    _Float16* Qh      = wT + (size_t)4 * WSZ;     // [h][b][l][dk]
    _Float16* Kh      = Qh + (size_t)NE;          // [h][b][t][dk]
    _Float16* VhT     = Kh + (size_t)NE;          // [h][b][dk][t]
    _Float16* attnout = VhT + (size_t)NE;         // [b*L][h*64+dk]

    k_cvt_w<<<(4 * WSZ + 255) / 256, 256, 0, stream>>>(wq, wk, wv, wfc, wT);
    k_proj<<<3 * 512 * 48, 32, 0, stream>>>(q, k, v, wT, bq, bk, bv, Qh, Kh, VhT);
    k_attn<<<BB * (LL / 16), 384, 0, stream>>>(Qh, Kh, VhT, ploc, kpm, wloc, bloc,
                                               fused, attnout);
    k_fc_ln<<<(BB * LL) / 16, 256, 0, stream>>>(attnout, wT + (size_t)3 * WSZ, bfc,
                                                q, lng, lnb, out);
}
